// GlobalANOQuantumCircuit_33595234189573
// MI455X (gfx1250) — compile-verified, matrix operand order checked by reading, not run on hardware
//
#include <hip/hip_runtime.h>
#include <hip/hip_bf16.h>

// ---------------------------------------------------------------------------
// Types for CDNA5 WMMA
// ---------------------------------------------------------------------------
typedef __attribute__((ext_vector_type(16))) __bf16 v16bf;
typedef __attribute__((ext_vector_type(8)))  float  v8f;
typedef __attribute__((ext_vector_type(2)))  float  v2f;

#if defined(__has_builtin)
#if __has_builtin(__builtin_amdgcn_wmma_f32_16x16x4_f32)
#define HAVE_WMMA_F32X4 1
#endif
#endif
#ifndef HAVE_WMMA_F32X4
#define HAVE_WMMA_F32X4 0
#endif

// CDNA5 async global->LDS copy (ASYNCcnt-tracked, no VGPR round trip).
// LDS operand = low 32 bits of the generic pointer (ISA 10.2: LDS aperture
// uses addr[31:0] as the LDS byte offset).
#define EXPM_ASYNC 1
#define ASYNC_LDS_LOAD_B32(ldsPtr, gPtr)                                      \
  asm volatile("global_load_async_to_lds_b32 %0, %1, off"                     \
               :: "v"((unsigned)(uintptr_t)(ldsPtr)), "v"(gPtr) : "memory")
#define WAIT_ASYNCCNT0() asm volatile("s_wait_asynccnt 0x0" ::: "memory")

__device__ __forceinline__ unsigned short f2bf(float f) {
  unsigned u = __builtin_bit_cast(unsigned, f);
  u += 0x7FFFu + ((u >> 16) & 1u);   // round-to-nearest-even
  return (unsigned short)(u >> 16);
}

// ---------------------------------------------------------------------------
// GEMM: C[M,N] = act(A[M,K] @ W[K,N] + bias)   (bf16 WMMA 16x16x32)
// Block: 512 threads = 16 waves. Block tile 128x128, K-chunk 64.
// Each wave owns a 32x32 region = 2x2 WMMA accumulators (register blocking:
// 8 WMMAs per barrier period, each fragment reused twice).
// ---------------------------------------------------------------------------
template <bool SILU>
__global__ __launch_bounds__(512)
void gemm_bias_act(const float* __restrict__ A, const float* __restrict__ W,
                   const float* __restrict__ bias, float* __restrict__ C,
                   int M, int N, int K, int lda, int ldw, int ldc) {
  __shared__ unsigned short As[128][72];  // [m][k] bf16 (padded stride)
  __shared__ unsigned short Ws[128][72];  // [n][k] bf16 (W transposed)

  const int tid  = threadIdx.x;
  const int lane = tid & 31;
  const int wave = tid >> 5;
  const int bm = blockIdx.y * 128;
  const int bn = blockIdx.x * 128;
  const int wr = wave >> 2, wc = wave & 3;      // wave grid 4x4
  const int row0 = wr * 32, col0 = wc * 32;
  const int l  = lane & 15, hi = lane >> 4;

  v8f acc00 = {}, acc01 = {}, acc10 = {}, acc11 = {};

  const int nChunk = (K + 63) >> 6;
  for (int kk = 0; kk < nChunk; ++kk) {
    const int k0 = kk * 64;
    // Stage A tile (128x64) -> bf16 LDS, coalesced on k. 16 elems/thread.
    #pragma unroll 4
    for (int e = 0; e < 16; ++e) {
      int idx = tid + e * 512;            // 0..8191
      int r = idx >> 6, c = idx & 63;
      int gk = k0 + c;
      float v = (gk < K) ? A[(size_t)(bm + r) * lda + gk] : 0.0f;
      As[r][c] = f2bf(v);
    }
    // Stage W tile (64x128), transposed into [n][k]; coalesced on n.
    #pragma unroll 4
    for (int e = 0; e < 16; ++e) {
      int idx = tid + e * 512;
      int kr = idx >> 7, nc = idx & 127;
      int gk = k0 + kr, gn = bn + nc;
      float v = (gk < K && gn < N) ? W[(size_t)gk * ldw + gn] : 0.0f;
      Ws[nc][kr] = f2bf(v);
    }
    // Prefetch a slice of the next W chunk into cache (global_prefetch_b8).
    if (k0 + 64 < K) {
      int kr = tid >> 7, nc = tid & 127;
      int gk = k0 + 64 + kr, gn = bn + nc;
      if (gk < K && gn < N) __builtin_prefetch(&W[(size_t)gk * ldw + gn], 0, 1);
    }
    __syncthreads();

    #pragma unroll
    for (int ks = 0; ks < 64; ks += 32) {
      // Fragments per ISA 16-bit A 16x32 layout (B mirrors via W^T rows).
      union { v16bf v; unsigned short s[16]; } a0, a1, b0, b1;
      #pragma unroll
      for (int e = 0; e < 8; ++e) {
        a0.s[e]     = As[row0 + l][ks + hi * 8 + e];
        a0.s[e + 8] = As[row0 + l][ks + 16 + hi * 8 + e];
        a1.s[e]     = As[row0 + 16 + l][ks + hi * 8 + e];
        a1.s[e + 8] = As[row0 + 16 + l][ks + 16 + hi * 8 + e];
        b0.s[e]     = Ws[col0 + l][ks + hi * 8 + e];
        b0.s[e + 8] = Ws[col0 + l][ks + 16 + hi * 8 + e];
        b1.s[e]     = Ws[col0 + 16 + l][ks + hi * 8 + e];
        b1.s[e + 8] = Ws[col0 + 16 + l][ks + 16 + hi * 8 + e];
      }
      acc00 = __builtin_amdgcn_wmma_f32_16x16x32_bf16(false, a0.v, false, b0.v, (short)0, acc00, false, false);
      acc01 = __builtin_amdgcn_wmma_f32_16x16x32_bf16(false, a0.v, false, b1.v, (short)0, acc01, false, false);
      acc10 = __builtin_amdgcn_wmma_f32_16x16x32_bf16(false, a1.v, false, b0.v, (short)0, acc10, false, false);
      acc11 = __builtin_amdgcn_wmma_f32_16x16x32_bf16(false, a1.v, false, b1.v, (short)0, acc11, false, false);
    }
    __syncthreads();
  }

  // Epilogue: bias (+ SiLU), guarded on N edge.
  #pragma unroll
  for (int v = 0; v < 2; ++v) {
    int gn = bn + col0 + v * 16 + l;
    if (gn >= N) continue;
    float bv = bias[gn];
    #pragma unroll
    for (int u = 0; u < 2; ++u) {
      v8f acc = (u == 0) ? (v == 0 ? acc00 : acc01) : (v == 0 ? acc10 : acc11);
      #pragma unroll
      for (int r = 0; r < 8; ++r) {
        int gm = bm + row0 + u * 16 + hi * 8 + r;
        float val = acc[r] + bv;
        if (SILU) val = val / (1.0f + __expf(-val));
        C[(size_t)gm * ldc + gn] = val;
      }
    }
  }
}

// ---------------------------------------------------------------------------
// Pauli index/phase table: entry(x,z) = g4 | (q<<12),
//   g4 = base-4 index of Pauli string (0 == identity), q = popc(x&z) mod 4.
// P[i, i^x] = i^q * (-1)^popc(z & (i^x)),  digit map (x,z)->{I,X,Y,Z}.
// ---------------------------------------------------------------------------
__global__ void build_pauli_table(unsigned short* __restrict__ tab) {
  int x = blockIdx.x;     // 0..63
  int z = threadIdx.x;    // 0..63
  int g4 = 0;
  #pragma unroll
  for (int b = 0; b < 6; ++b) {
    int xb = (x >> b) & 1, zb = (z >> b) & 1;
    int p = xb ? (zb ? 2 : 1) : (zb ? 3 : 0);   // I=0 X=1 Y=2 Z=3
    g4 |= p << (2 * b);                          // qubit k <-> bit (5-k)
  }
  int q = __popc(x & z) & 3;
  tab[x * 64 + z] = (unsigned short)(g4 | (q << 12));
}

// ---------------------------------------------------------------------------
// Build G = i * (sum_g theta[b,g] * gen_g) using Pauli sparsity.
// Output layout: G[b][0][64][64] = Re, G[b][1][64][64] = Im.
// ---------------------------------------------------------------------------
__global__ __launch_bounds__(256)
void build_g(const float* __restrict__ theta, const unsigned short* __restrict__ tab,
             float* __restrict__ G, int ldt) {
  __shared__ float          th[4096];   // th[0]=0 handles the skipped identity
  __shared__ unsigned short t2[4096];
  const int b = blockIdx.x, tid = threadIdx.x;
  for (int e = tid; e < 4096; e += 256) {
    th[e] = (e == 0) ? 0.0f : theta[(size_t)b * ldt + (e - 1)];
    t2[e] = tab[e];
  }
  __syncthreads();
  for (int e = tid; e < 4096; e += 256) {
    int i = e >> 6, j = e & 63;
    int x = i ^ j;
    const unsigned short* row = &t2[x * 64];
    float re = 0.0f, im = 0.0f;
    for (int z = 0; z < 64; ++z) {
      unsigned short pk = row[z];
      float t = th[pk & 4095];
      float sg = (__popc(z & j) & 1) ? -t : t;
      switch (pk >> 12) {                 // i^q
        case 0: re += sg; break;
        case 1: im += sg; break;
        case 2: re -= sg; break;
        default: im -= sg; break;
      }
    }
    // multiply by i:  (re + i*im) * i = -im + i*re
    G[((size_t)b * 2 + 0) * 4096 + e] = -im;
    G[((size_t)b * 2 + 1) * 4096 + e] = re;
  }
}

// ---------------------------------------------------------------------------
// expm via scaling-and-squaring + 12-term Horner Taylor, complex 64x64.
// Matrices live in dynamic LDS; complex matmul = 4x fp32 WMMA 16x16x4.
// G is staged into LDS with CDNA5 async global->LDS loads (ASYNCcnt).
// ---------------------------------------------------------------------------
#define ELD 66  // padded LDS row stride (floats)

__device__ __forceinline__ void cmatmul64(
    const float* __restrict__ Are, const float* __restrict__ Aim,
    const float* __restrict__ Bre, const float* __restrict__ Bim,
    float* __restrict__ Dre, float* __restrict__ Dim,
    float scale, bool addI, int tid) {
  const int lane = tid & 31, wave = tid >> 5;
  const int ti = wave >> 2, tj = wave & 3;
  const int l = lane & 15, hi = lane >> 4;
  v8f cre = {}, cim = {};
#if HAVE_WMMA_F32X4
  const int m = ti * 16 + l;
  const int n = tj * 16 + l;
  #pragma unroll 4
  for (int kk = 0; kk < 16; ++kk) {
    int kb = kk * 4 + hi * 2;
    v2f are  = { Are[m * ELD + kb], Are[m * ELD + kb + 1] };
    v2f aim  = { Aim[m * ELD + kb], Aim[m * ELD + kb + 1] };
    v2f naim = { -aim.x, -aim.y };   // ISA: NEG[0:1] must be 0 for float WMMA
    v2f bre  = { Bre[kb * ELD + n], Bre[(kb + 1) * ELD + n] };
    v2f bim  = { Bim[kb * ELD + n], Bim[(kb + 1) * ELD + n] };
    cre = __builtin_amdgcn_wmma_f32_16x16x4_f32(false, are,  false, bre, (short)0, cre, false, false);
    cre = __builtin_amdgcn_wmma_f32_16x16x4_f32(false, naim, false, bim, (short)0, cre, false, false);
    cim = __builtin_amdgcn_wmma_f32_16x16x4_f32(false, are,  false, bim, (short)0, cim, false, false);
    cim = __builtin_amdgcn_wmma_f32_16x16x4_f32(false, aim,  false, bre, (short)0, cim, false, false);
  }
  #pragma unroll
  for (int r = 0; r < 8; ++r) {
    int row = ti * 16 + hi * 8 + r;
    int col = tj * 16 + l;
    float vr = cre[r] * scale, vi = cim[r] * scale;
    if (addI && row == col) vr += 1.0f;
    Dre[row * ELD + col] = vr;
    Dim[row * ELD + col] = vi;
  }
#else
  for (int e = tid; e < 4096; e += 512) {
    int i = e >> 6, j = e & 63;
    float vr = 0.0f, vi = 0.0f;
    for (int k = 0; k < 64; ++k) {
      float ar = Are[i * ELD + k], ai = Aim[i * ELD + k];
      float br = Bre[k * ELD + j], bi = Bim[k * ELD + j];
      vr = __fmaf_rn(ar, br, __fmaf_rn(-ai, bi, vr));
      vi = __fmaf_rn(ar, bi, __fmaf_rn( ai, br, vi));
    }
    vr *= scale; vi *= scale;
    if (addI && i == j) vr += 1.0f;
    Dre[i * ELD + j] = vr;
    Dim[i * ELD + j] = vi;
  }
#endif
}

__global__ __launch_bounds__(512)
void expm_kernel(const float* __restrict__ G, float* __restrict__ psi) {
  extern __shared__ float smem[];
  const int S = 64 * ELD;
  float* Tre  = smem;           float* Tim  = smem + S;
  float* P0re = smem + 2 * S;   float* P0im = smem + 3 * S;
  float* P1re = smem + 4 * S;   float* P1im = smem + 5 * S;
  float* red  = smem + 6 * S;   // 64 floats
  __shared__ int s_sh;

  const int b = blockIdx.x, tid = threadIdx.x;
  const float* Gre = G + (size_t)b * 8192;
  const float* Gim = Gre + 4096;

#if EXPM_ASYNC
  // Async DMA of G into LDS (16 async ops per wave, ASYNCcnt-tracked).
  for (int e = tid; e < 4096; e += 512) {
    int i = e >> 6, j = e & 63;
    ASYNC_LDS_LOAD_B32(&Tre[i * ELD + j], &Gre[e]);
    ASYNC_LDS_LOAD_B32(&Tim[i * ELD + j], &Gim[e]);
  }
  WAIT_ASYNCCNT0();
#else
  for (int e = tid; e < 4096; e += 512) {
    Tre[(e >> 6) * ELD + (e & 63)] = Gre[e];
    Tim[(e >> 6) * ELD + (e & 63)] = Gim[e];
  }
#endif
  __syncthreads();

  // 1-norm-ish bound: max row sum of |Re|+|Im|
  if (tid < 64) {
    float s = 0.0f;
    for (int j = 0; j < 64; ++j)
      s += fabsf(Tre[tid * ELD + j]) + fabsf(Tim[tid * ELD + j]);
    red[tid] = s;
  }
  __syncthreads();
  if (tid == 0) {
    float nm = 0.0f;
    for (int r = 0; r < 64; ++r) nm = fmaxf(nm, red[r]);
    int s = 0;
    if (nm > 0.25f) s = (int)ceilf(log2f(nm * 4.0f));
    s_sh = (s < 0) ? 0 : (s > 40 ? 40 : s);
  }
  __syncthreads();
  const int s = s_sh;
  const float scl = exp2f((float)-s);
  for (int e = tid; e < 4096; e += 512) {
    Tre[(e >> 6) * ELD + (e & 63)] *= scl;
    Tim[(e >> 6) * ELD + (e & 63)] *= scl;
  }
  for (int e = tid; e < 4096; e += 512) {   // P0 = I
    int i = e >> 6, j = e & 63;
    P0re[i * ELD + j] = (i == j) ? 1.0f : 0.0f;
    P0im[i * ELD + j] = 0.0f;
  }
  __syncthreads();

  int cur = 0;
  // Horner:  P <- I + (T/k) @ P,   k = 12..1   =>  P ~= exp(T)
  for (int k = 12; k >= 1; --k) {
    float* sre = cur ? P1re : P0re; float* sim = cur ? P1im : P0im;
    float* dre = cur ? P0re : P1re; float* dim = cur ? P0im : P1im;
    cmatmul64(Tre, Tim, sre, sim, dre, dim, 1.0f / (float)k, true, tid);
    cur ^= 1;
    __syncthreads();
  }
  // Squaring: P <- P @ P, s times
  for (int t = 0; t < s; ++t) {
    float* sre = cur ? P1re : P0re; float* sim = cur ? P1im : P0im;
    float* dre = cur ? P0re : P1re; float* dim = cur ? P0im : P1im;
    cmatmul64(sre, sim, sre, sim, dre, dim, 1.0f, false, tid);
    cur ^= 1;
    __syncthreads();
  }
  // psi = U[:,0]
  if (tid < 64) {
    float* pre = cur ? P1re : P0re; float* pim = cur ? P1im : P0im;
    psi[(size_t)b * 128 + tid]      = pre[tid * ELD + 0];
    psi[(size_t)b * 128 + 64 + tid] = pim[tid * ELD + 0];
  }
}

// ---------------------------------------------------------------------------
// Variational circuit: DEPTH x (RY on each qubit, then CNOT ring).
// qubit q <-> bit (5-q) of the amplitude index.
// ---------------------------------------------------------------------------
__global__ __launch_bounds__(64)
void circuit_kernel(const float* __restrict__ vp, float* __restrict__ psi) {
  __shared__ float sre[64], sim[64];
  const int b = blockIdx.x, t = threadIdx.x;
  float pr = psi[(size_t)b * 128 + t];
  float pi = psi[(size_t)b * 128 + 64 + t];
  for (int d = 0; d < 4; ++d) {
    for (int q = 0; q < 6; ++q) {
      float th = vp[d * 6 + q] * 0.5f;
      float c = cosf(th), sn = sinf(th);
      int bit = 1 << (5 - q);
      sre[t] = pr; sim[t] = pi;
      __syncthreads();
      float r0 = sre[t & ~bit], i0 = sim[t & ~bit];
      float r1 = sre[t |  bit], i1 = sim[t |  bit];
      if ((t & bit) == 0) { pr = c * r0 - sn * r1; pi = c * i0 - sn * i1; }
      else                { pr = sn * r0 + c * r1; pi = sn * i0 + c * i1; }
      __syncthreads();
    }
    for (int q = 0; q < 6; ++q) {
      int cb = 1 << (5 - q);
      int tb = 1 << (5 - ((q + 1) % 6));
      sre[t] = pr; sim[t] = pi;
      __syncthreads();
      int src = (t & cb) ? (t ^ tb) : t;
      pr = sre[src]; pi = sim[src];
      __syncthreads();
    }
  }
  psi[(size_t)b * 128 + t]      = pr;
  psi[(size_t)b * 128 + 64 + t] = pi;
}

// ---------------------------------------------------------------------------
// exp[b,w] = sum_i 2*D[w,i+1]*|psi_i|^2 (i<63)
//          + sum_{i>j} 2*( A*(ri*rj+ii*ij) + B*(ii*rj-ri*ij) )
// ---------------------------------------------------------------------------
__global__ __launch_bounds__(256)
void expect_kernel(const float* __restrict__ psi, const float* __restrict__ A,
                   const float* __restrict__ Bm, const float* __restrict__ D,
                   float* __restrict__ out) {
  __shared__ float sre[64], sim[64], red[256];
  const int w = blockIdx.x, b = blockIdx.y, tid = threadIdx.x;
  if (tid < 64) {
    sre[tid] = psi[(size_t)b * 128 + tid];
    sim[tid] = psi[(size_t)b * 128 + 64 + tid];
  }
  __syncthreads();
  float acc = 0.0f;
  for (int p = tid; p < 2016; p += 256) {
    int i = (int)((1.0f + sqrtf(1.0f + 8.0f * (float)p)) * 0.5f);
    while (i * (i - 1) / 2 > p) --i;
    while ((i + 1) * i / 2 <= p) ++i;
    int j = p - i * (i - 1) / 2;
    float a = A[w * 2016 + p], bb = Bm[w * 2016 + p];
    float ri = sre[i], ii = sim[i], rj = sre[j], ij = sim[j];
    acc += 2.0f * (a * (ri * rj + ii * ij) + bb * (ii * rj - ri * ij));
  }
  if (tid < 63) {
    float d = D[w * 64 + tid + 1];
    acc += 2.0f * d * (sre[tid] * sre[tid] + sim[tid] * sim[tid]);
  }
  red[tid] = acc;
  __syncthreads();
  for (int st = 128; st > 0; st >>= 1) {
    if (tid < st) red[tid] += red[tid + st];
    __syncthreads();
  }
  if (tid == 0) out[b * 15 + w] = red[0];
}

// ---------------------------------------------------------------------------
// Launch
// ---------------------------------------------------------------------------
extern "C" void kernel_launch(void* const* d_in, const int* in_sizes, int n_in,
                              void* d_out, int out_size, void* d_ws, size_t ws_size,
                              hipStream_t stream) {
  const float* x  = (const float*)d_in[0];
  const float* W1 = (const float*)d_in[1];
  const float* b1 = (const float*)d_in[2];
  const float* W2 = (const float*)d_in[3];
  const float* b2 = (const float*)d_in[4];
  const float* vp = (const float*)d_in[5];
  const float* A  = (const float*)d_in[6];
  const float* Bm = (const float*)d_in[7];
  const float* D  = (const float*)d_in[8];
  float* out = (float*)d_out;

  float* ws    = (float*)d_ws;
  float* h     = ws;                        // 256 * 2048
  float* theta = h + 256 * 2048;            // 256 * 4096
  float* G     = theta + 256 * 4096;        // 256 * 8192 (re/im planes)
  float* psi   = G + (size_t)256 * 8192;    // 256 * 128
  unsigned short* tab = (unsigned short*)(psi + 256 * 128);  // 4096 u16

  // (1) h = silu(x @ W1 + b1):  M=256 N=2047 K=1024
  gemm_bias_act<true><<<dim3(16, 2), 512, 0, stream>>>(
      x, W1, b1, h, 256, 2047, 1024, 1024, 2047, 2048);
  // (2) theta = h @ W2 + b2:    M=256 N=4095 K=2047
  gemm_bias_act<false><<<dim3(32, 2), 512, 0, stream>>>(
      h, W2, b2, theta, 256, 4095, 2047, 2048, 4095, 4096);
  // (3) Pauli phase table + G = i * sum_g theta_g gen_g (sparse structure)
  build_pauli_table<<<64, 64, 0, stream>>>(tab);
  build_g<<<256, 256, 0, stream>>>(theta, tab, G, 4096);
  // (4) U = expm(G); psi = U[:,0]
  size_t expm_lds = (size_t)(6 * 64 * ELD + 64) * sizeof(float);
  expm_kernel<<<256, 512, expm_lds, stream>>>(G, psi);
  // (5) variational circuit
  circuit_kernel<<<256, 64, 0, stream>>>(vp, psi);
  // (6) expectation values
  expect_kernel<<<dim3(15, 256), 256, 0, stream>>>(psi, A, Bm, D, out);
}
